// TreatmentConditionedAttention_64355789963688
// MI455X (gfx1250) — matrix operand order repaired and checked by reading, not compiled, not run on hardware
//
#include <hip/hip_runtime.h>
#include <hip/hip_bf16.h>

#define HIDDEN  1024
#define NHEADS  16
#define HEADDIM 64
#define NBATCH  4
#define NSEQ    2048

typedef __attribute__((ext_vector_type(16))) __bf16 v16bf;
typedef __attribute__((ext_vector_type(8)))  __bf16 v8bf;
typedef __attribute__((ext_vector_type(8)))  float  v8f;

union bfrag { v16bf v; v8bf h[2]; };

#if defined(__has_builtin)
#if __has_builtin(__builtin_amdgcn_global_load_async_to_lds_b128)
#define HAVE_ASYNC_LDS 1
#endif
#endif

#ifdef HAVE_ASYNC_LDS
typedef int v4i_g __attribute__((vector_size(16)));
typedef __attribute__((address_space(1))) v4i_g* as1_v4i_ptr;
typedef __attribute__((address_space(3))) v4i_g* as3_v4i_ptr;
#endif

// 16-byte global -> LDS copy; async DMA (ASYNCcnt) when available.
__device__ __forceinline__ void async_copy16(const __bf16* g, __bf16* l)
{
#ifdef HAVE_ASYNC_LDS
    __builtin_amdgcn_global_load_async_to_lds_b128(
        (as1_v4i_ptr)g, (as3_v4i_ptr)l, 0, 0);
#else
    *(v8bf*)l = *(const v8bf*)g;
#endif
}

__device__ __forceinline__ void wait_async_le4()
{
#if defined(__has_builtin) && __has_builtin(__builtin_amdgcn_s_wait_asynccnt)
    __builtin_amdgcn_s_wait_asynccnt(4);
#elif defined(HAVE_ASYNC_LDS)
    asm volatile("s_wait_asynccnt 0x4" ::: "memory");
#endif
}

__device__ __forceinline__ void wait_async_le0()
{
#if defined(__has_builtin) && __has_builtin(__builtin_amdgcn_s_wait_asynccnt)
    __builtin_amdgcn_s_wait_asynccnt(0);
#elif defined(HAVE_ASYNC_LDS)
    asm volatile("s_wait_asynccnt 0x0" ::: "memory");
#endif
}

__device__ __forceinline__ float eluf(float x) { return x > 0.f ? x : expm1f(x); }

// ---------------------------------------------------------------------------
// Kernel 1: per-batch conditioning vectors.
//   kmod[b][1024]  = cancer_kmod_emb[ct] + ELU(g@GkW+b) + ELU(t@TkW+b)
//   biasBH[b][16]  = cancer_bias_emb[ct] + MLP(g) + MLP(t)
// ---------------------------------------------------------------------------
__global__ __launch_bounds__(256) void prep_mods(
    const float* __restrict__ graph, const float* __restrict__ treat,
    const int*   __restrict__ ctype,
    const float* __restrict__ kmodEmb, const float* __restrict__ biasEmb,
    const float* __restrict__ GkW, const float* __restrict__ Gkb,
    const float* __restrict__ TkW, const float* __restrict__ Tkb,
    const float* __restrict__ GbW1, const float* __restrict__ Gbb1,
    const float* __restrict__ GbW2, const float* __restrict__ Gbb2,
    const float* __restrict__ TbW1, const float* __restrict__ Tbb1,
    const float* __restrict__ TbW2, const float* __restrict__ Tbb2,
    float* __restrict__ kmod, float* __restrict__ biasBH)
{
    int b = blockIdx.x;
    int t = threadIdx.x;
    __shared__ float g[256];
    __shared__ float tr[128];
    __shared__ float h1[64];
    g[t] = graph[b * 256 + t];
    if (t < 128) tr[t] = treat[b * 128 + t];
    __syncthreads();
    int ct = ctype[b];

    for (int j = t; j < HIDDEN; j += 256) {
        float sg = Gkb[j], st = Tkb[j];
        for (int i = 0; i < 256; ++i) sg += g[i] * GkW[i * HIDDEN + j];
        for (int i = 0; i < 128; ++i) st += tr[i] * TkW[i * HIDDEN + j];
        kmod[b * HIDDEN + j] = kmodEmb[ct * HIDDEN + j] + eluf(sg) + eluf(st);
    }

    if (t < 32) {
        float s = Gbb1[t];
        for (int i = 0; i < 256; ++i) s += g[i] * GbW1[i * 32 + t];
        h1[t] = eluf(s);
    } else if (t < 64) {
        int j = t - 32;
        float s = Tbb1[j];
        for (int i = 0; i < 128; ++i) s += tr[i] * TbW1[i * 32 + j];
        h1[t] = eluf(s);
    }
    __syncthreads();
    if (t < 16) {
        float sg = Gbb2[t], st = Tbb2[t];
        for (int i = 0; i < 32; ++i) {
            sg += h1[i]      * GbW2[i * 16 + t];
            st += h1[32 + i] * TbW2[i * 16 + t];
        }
        biasBH[b * 16 + t] = biasEmb[ct * 16 + t] + sg + st;
    }
}

// ---------------------------------------------------------------------------
// Kernel 2: fp32 -> bf16 elementwise
// ---------------------------------------------------------------------------
__global__ __launch_bounds__(256) void cvt_bf16(const float* __restrict__ src,
                                                __bf16* __restrict__ dst, int n)
{
    for (int i = blockIdx.x * blockDim.x + threadIdx.x; i < n;
         i += gridDim.x * blockDim.x)
        dst[i] = (__bf16)src[i];
}

// ---------------------------------------------------------------------------
// Kernel 3: W[k][n] fp32 -> Wt[n][k] bf16 (LDS tile transpose)
// ---------------------------------------------------------------------------
__global__ __launch_bounds__(256) void transpose_to_bf16(
    const float* __restrict__ W, __bf16* __restrict__ Wt)
{
    const int K = HIDDEN, N = HIDDEN;
    __shared__ float tile[32][33];
    int n0 = blockIdx.x * 32, k0 = blockIdx.y * 32;
    int tx = threadIdx.x, ty = threadIdx.y;           // 32 x 8
    for (int i = 0; i < 32; i += 8)
        tile[ty + i][tx] = W[(size_t)(k0 + ty + i) * N + n0 + tx];
    __syncthreads();
    for (int i = 0; i < 32; i += 8)
        Wt[(size_t)(n0 + ty + i) * K + k0 + tx] = (__bf16)tile[tx][ty + i];
}

// ---------------------------------------------------------------------------
// Kernel 4: bf16 WMMA GEMM with double-buffered async global->LDS staging.
//   C[M,1024] = A[M,1024] @ Wt^T (+bias)
//   A row-major bf16, Wt row-major [out][in] bf16.
//   Block tile 128x128, 8 waves of 32x64, K-step 32.
//   Per K-step the block stages A(128x32) + B(128x32) = 16 KB into LDS via
//   GLOBAL_LOAD_ASYNC_TO_LDS_B128 (4 instr/wave), overlapped with WMMAs on
//   the other buffer; fragments then come from LDS (pitch 40 elems = 20
//   DWORDs -> the 16 row-lanes land on distinct banks).
//   __launch_bounds__(256, 1): min 1 wave/EU so the ~150-VGPR working set
//   (64 accumulator + 48 fragment + addressing) stays register-resident —
//   the default occupancy floor forced accumulator spills to scratch.
//   mode 0: bf16 row-major out (Q)
//   mode 1: bf16 row-major out + kmod[row>>11] add (K)
//   mode 2: bf16 transposed per-head out Vt[b][h][d][tok] (V)
//   mode 3: fp32 row-major out (final projection)
// ---------------------------------------------------------------------------
__global__ __launch_bounds__(256, 1) void gemm_bf16(
    const __bf16* __restrict__ A, const __bf16* __restrict__ Wt,
    const float* __restrict__ bias, const float* __restrict__ kmod,
    void* __restrict__ out, int mode)
{
    const int K = HIDDEN;
    constexpr int LDK = 40;                       // padded k pitch (elements)
    __shared__ __align__(16) __bf16 As[2][128][LDK];
    __shared__ __align__(16) __bf16 Bs[2][128][LDK];

    int tid  = threadIdx.x;
    int wave = tid >> 5, lane = tid & 31;
    int wm = wave & 3, wn = wave >> 2;            // 4 x 2 wave grid
    int m0 = blockIdx.x * 128;
    int n0 = blockIdx.y * 128;
    int half = lane >> 4, l16 = lane & 15;

    // staging: each thread moves 2x16B of A and 2x16B of B per K-step
    int srow = tid >> 1;                          // 0..127
    int scol = (tid & 1) * 16;                    // element offset 0 / 16
    const __bf16* agsrc = A  + (size_t)(m0 + srow) * K + scol;
    const __bf16* bgsrc = Wt + (size_t)(n0 + srow) * K + scol;

    v8f acc[2][4] = {};

    auto stage = [&](int buf, int k0) {
        async_copy16(agsrc + k0,     &As[buf][srow][scol]);
        async_copy16(agsrc + k0 + 8, &As[buf][srow][scol + 8]);
        async_copy16(bgsrc + k0,     &Bs[buf][srow][scol]);
        async_copy16(bgsrc + k0 + 8, &Bs[buf][srow][scol + 8]);
    };

    stage(0, 0);
    int buf = 0;
    for (int k0 = 0; k0 < K; k0 += 32) {
        if (k0 + 32 < K) {
            stage(buf ^ 1, k0 + 32);   // prefetch next tile into other buffer
            wait_async_le4();          // oldest 4 (current tile) complete
        } else {
            wait_async_le0();
        }
        __syncthreads();

        bfrag af[2], bfr[4];
#pragma unroll
        for (int mt = 0; mt < 2; ++mt) {
            const __bf16* p = &As[buf][wm * 32 + mt * 16 + l16][half * 16];
            af[mt].h[0] = *(const v8bf*)p;
            af[mt].h[1] = *(const v8bf*)(p + 8);
        }
#pragma unroll
        for (int nt = 0; nt < 4; ++nt) {
            const __bf16* p = &Bs[buf][wn * 64 + nt * 16 + l16][half * 16];
            bfr[nt].h[0] = *(const v8bf*)p;
            bfr[nt].h[1] = *(const v8bf*)(p + 8);
        }
#pragma unroll
        for (int mt = 0; mt < 2; ++mt)
#pragma unroll
            for (int nt = 0; nt < 4; ++nt)
                acc[mt][nt] = __builtin_amdgcn_wmma_f32_16x16x32_bf16(
                    false, af[mt].v, false, bfr[nt].v,
                    (short)0, acc[mt][nt], false, false);

        __syncthreads();               // buffer free before it is re-staged
        buf ^= 1;
    }

#pragma unroll
    for (int mt = 0; mt < 2; ++mt) {
        int rowbase = m0 + wm * 32 + mt * 16 + half * 8;
#pragma unroll
        for (int nt = 0; nt < 4; ++nt) {
            int col = n0 + wn * 64 + nt * 16 + l16;
            float bv = bias[col];
#pragma unroll
            for (int r = 0; r < 8; ++r) {
                int row = rowbase + r;
                float v = acc[mt][nt][r] + bv;
                if (mode == 1) v += kmod[(row >> 11) * HIDDEN + col];
                if (mode == 2) {
                    int bb = row >> 11, tok = row & (NSEQ - 1);
                    int hh = col >> 6, dd = col & 63;
                    ((__bf16*)out)[(((size_t)bb * NHEADS + hh) * HEADDIM + dd) * NSEQ + tok] =
                        (__bf16)v;
                } else if (mode == 3) {
                    ((float*)out)[(size_t)row * HIDDEN + col] = v;
                } else {
                    ((__bf16*)out)[(size_t)row * HIDDEN + col] = (__bf16)v;
                }
            }
        }
    }
}

// ---------------------------------------------------------------------------
// Kernel 5: flash attention per (b,h).  Block = 128 query rows, 8 waves,
// each wave owns 16 rows; keys streamed in chunks of 32.
//   Q,K: [B][tok][h*64+d] bf16 row-major ; Vt: [B][H][d][tok] bf16
// ---------------------------------------------------------------------------
__global__ __launch_bounds__(256, 1) void attn_kernel(
    const __bf16* __restrict__ Q, const __bf16* __restrict__ Kk,
    const __bf16* __restrict__ Vt, const float* __restrict__ biasBH,
    __bf16* __restrict__ O16)
{
    int bh = blockIdx.y;
    int b = bh >> 4, h = bh & 15;
    int tid = threadIdx.x, wave = tid >> 5, lane = tid & 31;
    int half = lane >> 4, l16 = lane & 15;
    int qrow0 = blockIdx.x * 128 + wave * 16;

    __shared__ __align__(16) __bf16 Pbuf[8][16][48];   // per-wave 16x32 (+pad)

    // Q A-fragments, resident for the whole key loop (d is the WMMA-K dim)
    bfrag qf[2];
    {
        const __bf16* qp = Q + ((size_t)(b * NSEQ + qrow0 + l16)) * HIDDEN +
                           h * HEADDIM + half * 16;
#pragma unroll
        for (int kt = 0; kt < 2; ++kt) {
            qf[kt].h[0] = *(const v8bf*)(qp + kt * 32);
            qf[kt].h[1] = *(const v8bf*)(qp + kt * 32 + 8);
        }
    }
    float bias  = biasBH[bh];
    const float scale = 0.125f;       // 1/sqrt(64)

    float mrow[8], lrow[8];
    v8f oacc[4] = {};
#pragma unroll
    for (int r = 0; r < 8; ++r) { mrow[r] = -1e30f; lrow[r] = 0.f; }

    for (int kc = 0; kc < NSEQ; kc += 32) {
        // ---- S = Q @ K^T for 32 keys (2 pos-tiles x 2 d-chunks) ----
        v8f S[2] = {};
#pragma unroll
        for (int pt = 0; pt < 2; ++pt) {
            const __bf16* kp = Kk + ((size_t)(b * NSEQ + kc + pt * 16 + l16)) * HIDDEN +
                               h * HEADDIM + half * 16;
#pragma unroll
            for (int kt = 0; kt < 2; ++kt) {
                bfrag kf;
                kf.h[0] = *(const v8bf*)(kp + kt * 32);
                kf.h[1] = *(const v8bf*)(kp + kt * 32 + 8);
                S[pt] = __builtin_amdgcn_wmma_f32_16x16x32_bf16(
                    false, qf[kt].v, false, kf.v, (short)0, S[pt], false, false);
            }
        }
        // ---- online softmax update ----
#pragma unroll
        for (int r = 0; r < 8; ++r) {
            float s0 = S[0][r] * scale + bias;
            float s1 = S[1][r] * scale + bias;
            float c = fmaxf(s0, s1);
#pragma unroll
            for (int msk = 8; msk >= 1; msk >>= 1)
                c = fmaxf(c, __shfl_xor(c, msk, 16));
            float nm   = fmaxf(mrow[r], c);
            float corr = __expf(mrow[r] - nm);
            mrow[r] = nm;
            float p0 = __expf(s0 - nm);
            float p1 = __expf(s1 - nm);
            lrow[r] = lrow[r] * corr + p0 + p1;
#pragma unroll
            for (int dt = 0; dt < 4; ++dt) oacc[dt][r] *= corr;
            int m_ = r + 8 * half;
            Pbuf[wave][m_][l16]      = (__bf16)p0;
            Pbuf[wave][m_][16 + l16] = (__bf16)p1;
        }
        asm volatile("s_wait_dscnt 0" ::: "memory");   // wave-private LDS RAW
        // ---- O += P @ V ----
        bfrag pa;
        pa.h[0] = *(const v8bf*)&Pbuf[wave][l16][half * 16];
        pa.h[1] = *(const v8bf*)&Pbuf[wave][l16][half * 16 + 8];
#pragma unroll
        for (int dt = 0; dt < 4; ++dt) {
            const __bf16* vp = Vt + ((size_t)(bh * HEADDIM + dt * 16 + l16)) * NSEQ +
                               kc + half * 16;
            bfrag vf;
            vf.h[0] = *(const v8bf*)vp;
            vf.h[1] = *(const v8bf*)(vp + 8);
            oacc[dt] = __builtin_amdgcn_wmma_f32_16x16x32_bf16(
                false, pa.v, false, vf.v, (short)0, oacc[dt], false, false);
        }
    }

    // final normalize: reduce lrow across the 16 lanes of each half
#pragma unroll
    for (int r = 0; r < 8; ++r) {
        float s = lrow[r];
#pragma unroll
        for (int msk = 8; msk >= 1; msk >>= 1) s += __shfl_xor(s, msk, 16);
        lrow[r] = 1.0f / s;
    }
#pragma unroll
    for (int dt = 0; dt < 4; ++dt)
#pragma unroll
        for (int r = 0; r < 8; ++r) {
            float v = oacc[dt][r] * lrow[r];
            int tok = qrow0 + r + 8 * half;
            int col = h * HEADDIM + dt * 16 + l16;
            O16[((size_t)b * NSEQ + tok) * HIDDEN + col] = (__bf16)v;
        }
}

// ---------------------------------------------------------------------------
extern "C" void kernel_launch(void* const* d_in, const int* in_sizes, int n_in,
                              void* d_out, int out_size, void* d_ws, size_t ws_size,
                              hipStream_t stream)
{
    const float* x      = (const float*)d_in[0];
    const float* graph  = (const float*)d_in[1];
    const float* treat  = (const float*)d_in[2];
    const int*   ctype  = (const int*)  d_in[3];
    const float* Wq = (const float*)d_in[4];  const float* bq = (const float*)d_in[5];
    const float* Wk = (const float*)d_in[6];  const float* bk = (const float*)d_in[7];
    const float* Wv = (const float*)d_in[8];  const float* bv = (const float*)d_in[9];
    const float* Wo = (const float*)d_in[10]; const float* bo = (const float*)d_in[11];
    const float* biasEmb = (const float*)d_in[12];
    const float* kmodEmb = (const float*)d_in[13];
    const float* GbW1 = (const float*)d_in[14]; const float* Gbb1 = (const float*)d_in[15];
    const float* GbW2 = (const float*)d_in[16]; const float* Gbb2 = (const float*)d_in[17];
    const float* GkW  = (const float*)d_in[18]; const float* Gkb  = (const float*)d_in[19];
    const float* TkW  = (const float*)d_in[20]; const float* Tkb  = (const float*)d_in[21];
    const float* TbW1 = (const float*)d_in[22]; const float* Tbb1 = (const float*)d_in[23];
    const float* TbW2 = (const float*)d_in[24]; const float* Tbb2 = (const float*)d_in[25];

    char* ws = (char*)d_ws;
    const size_t M        = (size_t)NBATCH * NSEQ;            // 8192
    const size_t actBytes = M * HIDDEN * sizeof(__bf16);      // 16 MB
    const size_t wBytes   = (size_t)HIDDEN * HIDDEN * sizeof(__bf16); // 2 MB

    size_t off = 0;
    float*  kmod   = (float*)(ws + off);  off += NBATCH * HIDDEN * sizeof(float);
    float*  biasBH = (float*)(ws + off);  off += NBATCH * NHEADS * sizeof(float);
    off = (off + 255) & ~(size_t)255;
    __bf16* x16  = (__bf16*)(ws + off); off += actBytes;
    __bf16* wq16 = (__bf16*)(ws + off); off += wBytes;
    __bf16* wk16 = (__bf16*)(ws + off); off += wBytes;
    __bf16* wv16 = (__bf16*)(ws + off); off += wBytes;
    __bf16* wo16 = (__bf16*)(ws + off); off += wBytes;
    __bf16* q16  = (__bf16*)(ws + off); off += actBytes;
    __bf16* k16  = (__bf16*)(ws + off); off += actBytes;
    __bf16* v16t = (__bf16*)(ws + off); off += actBytes;
    __bf16* o16  = (__bf16*)(ws + off); off += actBytes;

    prep_mods<<<NBATCH, 256, 0, stream>>>(
        graph, treat, ctype, kmodEmb, biasEmb,
        GkW, Gkb, TkW, Tkb, GbW1, Gbb1, GbW2, Gbb2, TbW1, Tbb1, TbW2, Tbb2,
        kmod, biasBH);

    cvt_bf16<<<4096, 256, 0, stream>>>(x, x16, (int)(M * HIDDEN));

    dim3 tgrid(32, 32), tblk(32, 8);
    transpose_to_bf16<<<tgrid, tblk, 0, stream>>>(Wq, wq16);
    transpose_to_bf16<<<tgrid, tblk, 0, stream>>>(Wk, wk16);
    transpose_to_bf16<<<tgrid, tblk, 0, stream>>>(Wv, wv16);
    transpose_to_bf16<<<tgrid, tblk, 0, stream>>>(Wo, wo16);

    dim3 ggrid(64, 8);   // M/128 x N/128
    gemm_bf16<<<ggrid, 256, 0, stream>>>(x16, wq16, bq, nullptr, (void*)q16, 0);
    gemm_bf16<<<ggrid, 256, 0, stream>>>(x16, wk16, bk, kmod,    (void*)k16, 1);
    gemm_bf16<<<ggrid, 256, 0, stream>>>(x16, wv16, bv, nullptr, (void*)v16t, 2);

    attn_kernel<<<dim3(NSEQ / 128, NBATCH * NHEADS), 256, 0, stream>>>(
        q16, k16, v16t, biasBH, o16);

    gemm_bf16<<<ggrid, 256, 0, stream>>>(o16, wo16, bo, nullptr, d_out, 3);
}
